// SparseAttention_3899830305034
// MI455X (gfx1250) — compile-verified, matrix operand order checked by reading, not run on hardware
//
#include <hip/hip_runtime.h>
#include <hip/hip_bf16.h>

typedef __attribute__((ext_vector_type(16))) _Float16 v16h;
typedef __attribute__((ext_vector_type(8)))  float    v8f;

#define B_ 2
#define N_ 8192
#define D_ 256
#define K_ 512

// -------------------- 1. zero-fill sparse_w (537 MB, store-bound) ----------
__global__ void zero_kernel(float4* __restrict__ p, size_t n4) {
    size_t i = (size_t)blockIdx.x * blockDim.x + threadIdx.x;
    size_t stride = (size_t)gridDim.x * blockDim.x;
    float4 z = make_float4(0.f, 0.f, 0.f, 0.f);
    for (; i < n4; i += stride) p[i] = z;
}

// -------------------- 2. importance = K . w + b -> sortable u32 key --------
__global__ __launch_bounds__(128) void imp_kernel(const float* __restrict__ key,
                                                  const float* __restrict__ w,
                                                  const float* __restrict__ bimp,
                                                  unsigned* __restrict__ keys) {
    int wid = threadIdx.x >> 5, lane = threadIdx.x & 31;
    int row = blockIdx.x * 4 + wid;            // 0 .. B_*N_-1
    const float* kr = key + (size_t)row * D_;
    float s = 0.f;
#pragma unroll
    for (int j = 0; j < 8; ++j) s += kr[lane * 8 + j] * w[lane * 8 + j];
#pragma unroll
    for (int o = 16; o; o >>= 1) s += __shfl_xor(s, o, 32);
    s += bimp[0];
    if (lane == 0) {
        unsigned b = __float_as_uint(s);
        keys[row] = (b & 0x80000000u) ? ~b : (b | 0x80000000u);
    }
}

// -------------------- 3. exact top-512 set via radix select ----------------
__global__ __launch_bounds__(1024) void select_kernel(const unsigned* __restrict__ keys,
                                                      int* __restrict__ topk) {
    __shared__ unsigned u[N_];
    __shared__ int cnt, cg, ce;
    int tid = threadIdx.x;
    int batch = blockIdx.x;
    for (int i = tid; i < N_; i += 1024) u[i] = keys[batch * N_ + i];
    __syncthreads();

    unsigned prefix = 0;
    int kk = K_;
    for (int bit = 31; bit >= 0; --bit) {
        if (tid == 0) cnt = 0;
        __syncthreads();
        unsigned bm = 1u << bit;
        unsigned hi = ~((bm << 1) - 1u);       // bits above 'bit' (0 when bit==31)
        int local = 0;
        for (int i = tid; i < N_; i += 1024) {
            unsigned x = u[i];
            if ((x & hi) == prefix && (x & bm)) local++;
        }
        if (local) atomicAdd(&cnt, local);
        __syncthreads();
        int c = cnt;
        if (c >= kk) prefix |= bm; else kk -= c;
        __syncthreads();
    }
    int nG = K_ - kk;                          // # strictly greater than threshold
    if (tid == 0) { cg = 0; ce = 0; }
    __syncthreads();
    for (int i = tid; i < N_; i += 1024) {
        unsigned x = u[i];
        if (x > prefix) {
            int p = atomicAdd(&cg, 1);
            topk[batch * K_ + p] = i;
        } else if (x == prefix) {
            int p = atomicAdd(&ce, 1);
            if (p < kk) topk[batch * K_ + nG + p] = i;
        }
    }
}

// -------------------- 4. gather selected K/V rows as f16 -------------------
// ksel  : [b][k][d]  (row-major, A/B-frag K-runs contiguous for QK^T)
// vselT : [b][d][k]  (transposed, so PV B-frag K-runs are contiguous)
__global__ void gather_kernel(const float* __restrict__ key, const float* __restrict__ value,
                              const int* __restrict__ topk,
                              _Float16* __restrict__ ksel, _Float16* __restrict__ vselT) {
    int t = blockIdx.x * blockDim.x + threadIdx.x;   // 0 .. B_*K_*D_-1
    int bb  = t >> 17;                                // K_*D_ = 131072
    int rem = t & 131071;
    int kr  = rem >> 8;
    int c   = rem & 255;
    int src = topk[bb * K_ + kr];
    size_t off = ((size_t)(bb * N_ + src) << 8) + c;
    ksel[t] = (_Float16)key[off];
    vselT[((size_t)bb * D_ + c) * K_ + kr] = (_Float16)value[off];
}

// -------------------- 5. attention: WMMA QK^T, softmax, WMMA PV, scatter ---
// per-lane K-index pattern of a 16x32 f16 A/B fragment (ISA 7.12.2)
__device__ __forceinline__ int kpat(int hi16, int e) {
    return (hi16 << 3) + (e & 7) + ((e >> 3) << 4);
}

__global__ __launch_bounds__(128) void attn_kernel(const float* __restrict__ q,
                                                   const _Float16* __restrict__ kselh,
                                                   const _Float16* __restrict__ vselT,
                                                   const int* __restrict__ topk,
                                                   float* __restrict__ out,
                                                   float* __restrict__ sparse_w) {
    extern __shared__ char smem[];
    float* S  = (float*)smem;                       // [4 waves][16 rows][512]
    int*  tks = (int*)(smem + 4 * 16 * K_ * sizeof(float));

    int tid  = threadIdx.x;
    int wid  = tid >> 5, lane = tid & 31;
    int mrow = lane & 15, hi16 = lane >> 4;
    int batch = blockIdx.y;
    int qbase = blockIdx.x * 64 + wid * 16;

    for (int i = tid; i < K_; i += 128) tks[i] = topk[batch * K_ + i];
    __syncthreads();

    float* Sw = S + wid * 16 * K_;

    // ---- load Q tile as 8 A-fragments (K = 256) ----
    v16h aQ[8];
    const float* qrow = q + ((size_t)(batch * N_ + qbase + mrow)) * D_;
#pragma unroll
    for (int kb = 0; kb < 8; ++kb) {
#pragma unroll
        for (int e = 0; e < 16; ++e)
            aQ[kb][e] = (_Float16)qrow[kb * 32 + kpat(hi16, e)];
    }

    // ---- scores = Q . Ksel^T / 16 ----
    const float scale = 0.0625f;
    for (int kt = 0; kt < 32; ++kt) {
        v8f acc = {};
        const _Float16* krow = kselh + ((size_t)batch * K_ + kt * 16 + mrow) * D_;
#pragma unroll
        for (int kb = 0; kb < 8; ++kb) {
            v16h bf;
#pragma unroll
            for (int e = 0; e < 16; ++e)
                bf[e] = krow[kb * 32 + kpat(hi16, e)];
            acc = __builtin_amdgcn_wmma_f32_16x16x32_f16(false, aQ[kb], false, bf,
                                                         (short)0, acc, false, false);
        }
#pragma unroll
        for (int r = 0; r < 8; ++r)
            Sw[(r + (hi16 << 3)) * K_ + kt * 16 + mrow] = acc[r] * scale;
    }

    // ---- wave-local softmax over 512 keys per row ----
    for (int r = 0; r < 16; ++r) {
        float* row = Sw + r * K_;
        float m = -1e30f;
        for (int j = lane; j < K_; j += 32) m = fmaxf(m, row[j]);
#pragma unroll
        for (int o = 16; o; o >>= 1) m = fmaxf(m, __shfl_xor(m, o, 32));
        float s = 0.f;
        for (int j = lane; j < K_; j += 32) {
            float e = __expf(row[j] - m);
            row[j] = e;
            s += e;
        }
#pragma unroll
        for (int o = 16; o; o >>= 1) s += __shfl_xor(s, o, 32);
        float inv = 1.0f / s;
        for (int j = lane; j < K_; j += 32) row[j] *= inv;
    }

    // ---- out = attn . Vsel  (16 x 256 per wave) ----
    for (int nt = 0; nt < 16; ++nt) {
        v8f acc = {};
        // B-frags from transposed V: lane owns output column n = nt*16+mrow,
        // K-runs of 8 along k are contiguous -> b128 loads
        const _Float16* vrow = vselT + ((size_t)batch * D_ + nt * 16 + mrow) * K_;
#pragma unroll 4
        for (int kb = 0; kb < 16; ++kb) {
            v16h af, bf;
            const float* arow = Sw + mrow * K_ + kb * 32;
#pragma unroll
            for (int e = 0; e < 16; ++e)
                af[e] = (_Float16)arow[kpat(hi16, e)];
#pragma unroll
            for (int e = 0; e < 16; ++e)
                bf[e] = vrow[kb * 32 + kpat(hi16, e)];
            acc = __builtin_amdgcn_wmma_f32_16x16x32_f16(false, af, false, bf,
                                                         (short)0, acc, false, false);
        }
#pragma unroll
        for (int r = 0; r < 8; ++r)
            out[((size_t)(batch * N_ + qbase + r + (hi16 << 3))) * D_ + nt * 16 + mrow] = acc[r];
    }

    // ---- scatter attn weights into pre-zeroed sparse_w ----
    for (int idx = lane; idx < 16 * K_; idx += 32) {
        int r = idx >> 9, c = idx & (K_ - 1);
        sparse_w[((size_t)(batch * N_ + qbase + r)) * N_ + tks[c]] = Sw[r * K_ + c];
    }
}

extern "C" void kernel_launch(void* const* d_in, const int* in_sizes, int n_in,
                              void* d_out, int out_size, void* d_ws, size_t ws_size,
                              hipStream_t stream) {
    const float* q     = (const float*)d_in[0];
    const float* k     = (const float*)d_in[1];
    const float* v     = (const float*)d_in[2];
    const float* w_imp = (const float*)d_in[3];
    const float* b_imp = (const float*)d_in[4];
    (void)in_sizes; (void)n_in; (void)out_size; (void)ws_size;

    float* out = (float*)d_out;
    float* sw  = out + (size_t)B_ * N_ * D_;

    unsigned* keys  = (unsigned*)d_ws;                                    // B_*N_ u32
    int* topk       = (int*)((char*)d_ws + (size_t)B_ * N_ * 4);          // B_*K_ i32
    _Float16* ksel  = (_Float16*)((char*)topk + (size_t)B_ * K_ * 4);     // B_*K_*D_ f16
    _Float16* vselT = ksel + (size_t)B_ * K_ * D_;                        // B_*D_*K_ f16

    // 1. zero sparse output (store-bound: 537 MB)
    zero_kernel<<<4096, 256, 0, stream>>>((float4*)sw, (size_t)B_ * N_ * N_ / 4);
    // 2. importance scores
    imp_kernel<<<(B_ * N_) / 4, 128, 0, stream>>>(k, w_imp, b_imp, keys);
    // 3. top-512 per batch
    select_kernel<<<B_, 1024, 0, stream>>>(keys, topk);
    // 4. gather selected K rows (row-major) and V rows (transposed) as f16
    gather_kernel<<<(B_ * K_ * D_) / 256, 256, 0, stream>>>(k, v, topk, ksel, vselT);
    // 5. attention + scatter (128 KB LDS scores/WG -> dynamic LDS, 320 KB WGP budget)
    size_t shmem = 4 * 16 * K_ * sizeof(float) + K_ * sizeof(int);
    (void)hipFuncSetAttribute((const void*)attn_kernel,
                              hipFuncAttributeMaxDynamicSharedMemorySize, (int)shmem);
    attn_kernel<<<dim3(N_ / 64, B_), 128, shmem, stream>>>(q, ksel, vselT, topk, out, sw);
}
